// LSTM_MNIST_55937654063548
// MI455X (gfx1250) — matrix operand order repaired and checked by reading, not compiled, not run on hardware
//
#include <hip/hip_runtime.h>
#include <hip/hip_bf16.h>
#include <stdint.h>

// Problem constants (from reference)
#define BB 4096
#define TT 28
#define II 28
#define HH 300
#define OO 10
#define HP 320            // H padded to multiple of 32 (WMMA K and N-tiling)
#define GG 1200           // 4*H gate width (logical)
#define NTILES 80         // 4 gate types x 20 tiles of 16 (h padded to 320)
#define TILE_M 16         // batch rows per block
#define WAVES 5
#define THREADS (WAVES * 32)

typedef __attribute__((ext_vector_type(16))) _Float16 v16h;
typedef __attribute__((ext_vector_type(8)))  float    v8f;

union AFrag { v16h h; uint32_t u[8]; uint4 q[2]; };

__device__ __forceinline__ float fast_sigmoid(float x) {
    return 1.0f / (1.0f + __expf(-x));
}
__device__ __forceinline__ float fast_tanh(float x) {
    // 1 - 2/(e^{2x}+1); exp overflow/underflow saturates to +/-1 correctly
    return 1.0f - 2.0f / (__expf(2.0f * x) + 1.0f);
}

// ---------------------------------------------------------------------------
// Gate-permuted weight packing. Packed N-tile nt' in [0,80): type tau = nt'/20,
// h-tile ht = nt'%20; lane column h = ht*16 + (lane&15); original weight row
// N = tau*300 + h (zero if h >= 300).  Fragment (nt', kb) = 1KB, lane l holds
// K = kb*32 + (l>>4)*16 + e.  Wave w owns tiles { tau*20 + w*4 + j }, so each
// lane accumulates i,f,g,o of the SAME h -> whole cell update in registers.
// ---------------------------------------------------------------------------
__global__ void pack_w(const float* __restrict__ Wih, const float* __restrict__ Whh,
                       int in_dim, int in_pad, int KB, _Float16* __restrict__ dst) {
    int tid = blockIdx.x * blockDim.x + threadIdx.x;
    int total = NTILES * KB * 512;
    if (tid >= total) return;
    int frag = tid >> 9;
    int r    = tid & 511;
    int lane = r >> 4;
    int e    = r & 15;
    int nt = frag / KB, kb = frag % KB;
    int tau = nt / 20, ht = nt % 20;
    int h = ht * 16 + (lane & 15);
    int K = kb * 32 + ((lane >> 4) << 4) + e;
    float v = 0.0f;
    if (h < HH) {
        int N = tau * HH + h;
        if (K < in_pad) {
            if (K < in_dim) v = Wih[(size_t)N * in_dim + K];
        } else {
            int j = K - in_pad;
            if (j < HH) v = Whh[(size_t)N * HH + j];
        }
    }
    dst[tid] = (_Float16)v;
}

__global__ void bias_sum(const float* __restrict__ bih, const float* __restrict__ bhh,
                         float* __restrict__ dst) {
    int g = blockIdx.x * blockDim.x + threadIdx.x;
    if (g < GG) dst[g] = bih[g] + bhh[g];
}

// ---------------------------------------------------------------------------
// Fused LSTM layer. Block = 16 batch rows, 5 waves. Only LDS use: the shared
// A-row buffer [x_t | h_{t-1}] (f16, K padded to Kp <= 640, 20KB). Gates and
// cell state c live entirely in registers (gate-permuted packing).
// ---------------------------------------------------------------------------
template <int KB, int Kp, int IN_PAD, int IN_DIM, bool IS_L0>
__global__ __launch_bounds__(THREADS)
void lstm_layer_k(const void* __restrict__ in_seq,
                  const _Float16* __restrict__ wpk, const float* __restrict__ bias,
                  _Float16* __restrict__ out_seq) {
    __shared__ _Float16 sh_x[TILE_M * 640];

    const int tid  = threadIdx.x;
    const int lane = tid & 31;
    const int wv   = tid >> 5;
    const int b0   = blockIdx.x * TILE_M;
    const int m    = lane & 15;          // A-fragment row = batch-tile row
    const int half = lane >> 4;

    // h_{-1} = 0 and K-padding = 0
    for (int i = tid; i < TILE_M * Kp; i += THREADS) sh_x[i] = (_Float16)0.0f;

    // Per-lane h columns: hcol[j] = (wv*4 + j)*16 + (lane&15)
    float bc[4][4];
    #pragma unroll
    for (int ta = 0; ta < 4; ++ta)
        #pragma unroll
        for (int j = 0; j < 4; ++j) {
            int hj = (wv * 4 + j) * 16 + m;
            bc[ta][j] = (hj < HH) ? bias[ta * HH + hj] : 0.0f;
        }

    // Register-resident cell state: c[j][v] for row M = half*8+v, col hcol[j]
    float cst[4][8];
    #pragma unroll
    for (int j = 0; j < 4; ++j)
        #pragma unroll
        for (int v = 0; v < 8; ++v) cst[j][v] = 0.0f;

    // Weight base for this wave/lane; tile (tau,j) at constexpr offset.
    const char* wb = (const char*)wpk + (size_t)(wv * 4) * (KB * 1024) + (size_t)lane * 32;

    __syncthreads();

    for (int t = 0; t < TT; ++t) {
        // --- stage x_t into the A-row buffer -------------------------------
        if (IS_L0) {
            const float* xin = (const float*)in_seq;
            for (int i = tid; i < TILE_M * IN_DIM; i += THREADS) {
                int r = i / IN_DIM, k = i % IN_DIM;
                sh_x[r * Kp + k] =
                    (_Float16)xin[((size_t)(b0 + r) * TT + t) * II + k];
            }
        } else {
            // 4 f16 per op; both src (HP=320) and dst (Kp) rows are 8B aligned
            const _Float16* xin = (const _Float16*)in_seq;
            for (int i = tid; i < TILE_M * (HH / 4); i += THREADS) {
                int r = i / (HH / 4), kq = i % (HH / 4);
                *(uint64_t*)&sh_x[r * Kp + kq * 4] =
                    *(const uint64_t*)&xin[((size_t)(b0 + r) * TT + t) * HP + kq * 4];
            }
        }
        __syncthreads();   // orders x-stage AND previous h-write before gemm

        // --- gates = [x_t|h] @ Wpk + (bih+bhh), f16 WMMA, f32 accum --------
        v8f acc[4][4];
        #pragma unroll
        for (int ta = 0; ta < 4; ++ta)
            #pragma unroll
            for (int j = 0; j < 4; ++j) {
                #pragma unroll
                for (int v = 0; v < 8; ++v) acc[ta][j][v] = bc[ta][j];
            }

        #pragma unroll 1
        for (int kb = 0; kb < KB; ++kb) {
            // A fragment: two 16B-aligned LDS b128 loads (contiguous k = half*16 + e, same order as pack_w's B)
            AFrag a;
            const _Float16* ap = sh_x + m * Kp + kb * 32 + half * 16;
            a.q[0] = *(const uint4*)(ap);
            a.q[1] = *(const uint4*)(ap + 8);

            const char* p = wb + (size_t)kb * 1024;
            #pragma unroll
            for (int ta = 0; ta < 4; ++ta)
                #pragma unroll
                for (int j = 0; j < 4; ++j) {
                    v16h bf = *(const v16h*)(p + (ta * 20 + j) * (KB * 1024));
                    acc[ta][j] = __builtin_amdgcn_wmma_f32_16x16x32_f16(
                        false, a.h, false, bf, (short)0, acc[ta][j], false, false);
                }
        }
        __syncthreads();   // gemm's sh_x reads done before h_t overwrites it

        // --- cell update entirely in registers; broadcast h_t via LDS ------
        #pragma unroll
        for (int j = 0; j < 4; ++j) {
            int hj = (wv * 4 + j) * 16 + m;
            #pragma unroll
            for (int v = 0; v < 8; ++v) {
                float ig = fast_sigmoid(acc[0][j][v]);
                float fg = fast_sigmoid(acc[1][j][v]);
                float gg = fast_tanh(acc[2][j][v]);
                float og = fast_sigmoid(acc[3][j][v]);
                float c = fg * cst[j][v] + ig * gg;
                cst[j][v] = c;
                float hn = og * fast_tanh(c);
                int row = half * 8 + v;
                if (hj < HH) {
                    sh_x[row * Kp + IN_PAD + hj] = (_Float16)hn;   // next step's A
                    out_seq[((size_t)(b0 + row) * TT + t) * HP + hj] = (_Float16)hn;
                } else {
                    out_seq[((size_t)(b0 + row) * TT + t) * HP + hj] = (_Float16)0.0f;
                }
            }
        }
        // loop-top barrier orders these writes before the next gemm
    }
}

// ---------------------------------------------------------------------------
// out[b][o] = fcb[o] + sum_h h_last[b][h] * fcW[o][h]   (tiny: 0.025 GFLOP)
// ---------------------------------------------------------------------------
__global__ void fc_k(const _Float16* __restrict__ seq, const float* __restrict__ fcW,
                     const float* __restrict__ fcb, float* __restrict__ out) {
    int idx = blockIdx.x * blockDim.x + threadIdx.x;
    if (idx >= BB * OO) return;
    int b = idx / OO, o = idx % OO;
    const _Float16* h = seq + ((size_t)b * TT + (TT - 1)) * HP;
    float s = fcb[o];
    for (int j = 0; j < HH; ++j) s += (float)h[j] * fcW[o * HH + j];
    out[idx] = s;
}

// ---------------------------------------------------------------------------
// Workspace layout (bytes, 256B-aligned blocks)
// ---------------------------------------------------------------------------
static const size_t OFF_WPK0 = 0;                                 // 80*11*1024
static const size_t OFF_WPK1 = OFF_WPK0 + (size_t)NTILES * 11 * 1024;
static const size_t OFF_WPK2 = OFF_WPK1 + (size_t)NTILES * 20 * 1024;
static const size_t OFF_BSUM = OFF_WPK2 + (size_t)NTILES * 20 * 1024;
static const size_t OFF_SEQA = ((OFF_BSUM + 3 * GG * 4 + 255) / 256) * 256;
static const size_t SEQ_BYTES = (size_t)BB * TT * HP * 2;
static const size_t OFF_SEQB = OFF_SEQA + SEQ_BYTES;

extern "C" void kernel_launch(void* const* d_in, const int* in_sizes, int n_in,
                              void* d_out, int out_size, void* d_ws, size_t ws_size,
                              hipStream_t stream) {
    (void)in_sizes; (void)n_in; (void)out_size; (void)ws_size;

    const float* x = (const float*)d_in[0];
    const float* Wih[3] = {(const float*)d_in[1], (const float*)d_in[5], (const float*)d_in[9]};
    const float* Whh[3] = {(const float*)d_in[2], (const float*)d_in[6], (const float*)d_in[10]};
    const float* bih[3] = {(const float*)d_in[3], (const float*)d_in[7], (const float*)d_in[11]};
    const float* bhh[3] = {(const float*)d_in[4], (const float*)d_in[8], (const float*)d_in[12]};
    const float* fcW = (const float*)d_in[13];
    const float* fcb = (const float*)d_in[14];

    char* ws = (char*)d_ws;
    _Float16* wpk[3] = {(_Float16*)(ws + OFF_WPK0), (_Float16*)(ws + OFF_WPK1),
                        (_Float16*)(ws + OFF_WPK2)};
    float*    bsum   = (float*)(ws + OFF_BSUM);
    _Float16* seqA   = (_Float16*)(ws + OFF_SEQA);
    _Float16* seqB   = (_Float16*)(ws + OFF_SEQB);

    // Weight packing + bias fusion (f16 fragments, K padded: 352 / 640)
    const int KB0 = 11, KB12 = 20;
    pack_w<<<(NTILES * KB0 * 512 + 255) / 256, 256, 0, stream>>>(Wih[0], Whh[0], II, 32, KB0, wpk[0]);
    pack_w<<<(NTILES * KB12 * 512 + 255) / 256, 256, 0, stream>>>(Wih[1], Whh[1], HH, HP, KB12, wpk[1]);
    pack_w<<<(NTILES * KB12 * 512 + 255) / 256, 256, 0, stream>>>(Wih[2], Whh[2], HH, HP, KB12, wpk[2]);
    for (int l = 0; l < 3; ++l)
        bias_sum<<<(GG + 255) / 256, 256, 0, stream>>>(bih[l], bhh[l], bsum + (size_t)l * GG);

    // Three fused LSTM layers (ping-pong f16 sequence buffers)
    const int nblk = BB / TILE_M;   // 256
    lstm_layer_k<11, 352,  32, II, true ><<<nblk, THREADS, 0, stream>>>(x,    wpk[0], bsum + 0 * GG, seqA);
    lstm_layer_k<20, 640,  HP, HH, false><<<nblk, THREADS, 0, stream>>>(seqA, wpk[1], bsum + 1 * GG, seqB);
    lstm_layer_k<20, 640,  HP, HH, false><<<nblk, THREADS, 0, stream>>>(seqB, wpk[2], bsum + 2 * GG, seqA);

    // Final FC on h[:, T-1, :]
    fc_k<<<(BB * OO + 255) / 256, 256, 0, stream>>>(seqA, fcW, fcb, (float*)d_out);
}
